// DynSIHAMLP_28312424415700
// MI455X (gfx1250) — matrix operand order repaired and checked by reading, not compiled, not run on hardware
//
#include <hip/hip_runtime.h>
#include <hip/hip_bf16.h>
#include <stdint.h>

// Problem constants (from reference setup_inputs)
#define T_TOK   16384      // B*S
#define H_DIM   1024
#define E_EXP   16
#define F_DIM   4096
#define TOPK    2
#define MTILE   32         // tokens per FFN block (2 x WMMA M) -> halves weight re-reads

typedef __bf16 bf16_t;
typedef bf16_t v16bf __attribute__((ext_vector_type(16)));
typedef float  v8f   __attribute__((ext_vector_type(8)));

// typed pointers for the async global->LDS DMA builtin
typedef int v4i_ __attribute__((ext_vector_type(4)));
typedef __attribute__((address_space(1))) v4i_* as1_v4i_p;
typedef __attribute__((address_space(3))) v4i_* as3_v4i_p;

union BF16Tile { uint4 u4[2]; v16bf v; };

__device__ __forceinline__ unsigned short f2bf(float f) {
  unsigned int u = __float_as_uint(f);
  u += 0x7FFFu + ((u >> 16) & 1u);           // round-to-nearest-even
  return (unsigned short)(u >> 16);
}

// A operand (16x32 bf16, M x K) from LDS, row-major with row stride `ld`.
// ISA 7.12.2: lanes 0-15 -> M=lane, VGPR v<4: K=2v..2v+1 ; v>=4: K=16+2(v-4)..;
// lanes 16-31 get K offset +8. 16 bf16 per lane = two 16B LDS loads.
__device__ __forceinline__ v16bf ldsA(const unsigned short* lds, int ld, int kbase) {
  const int lane = threadIdx.x & 31;
  const int m = lane & 15, lh = lane >> 4;
  const unsigned short* p = lds + m * ld + kbase + 8 * lh;
  BF16Tile r;
  r.u4[0] = *(const uint4*)(p);        // k = kbase+8lh .. +7
  r.u4[1] = *(const uint4*)(p + 16);   // k = kbase+16+8lh .. +7
  return r.v;
}

// B operand (32x16 bf16, K x N) from global. Weight stored transposed
// as [N][K] (K contiguous, row stride ldk) so each lane reads two 16B runs:
// lanes 0-15 -> n=lane, K=0..7,16..23; lanes 16-31 -> same n, K=8..15,24..31.
__device__ __forceinline__ v16bf glbB(const unsigned short* g, size_t ldk,
                                      int nbase, int kbase) {
  const int lane = threadIdx.x & 31;
  const int n = lane & 15, lh = lane >> 4;
  const unsigned short* p = g + (size_t)(nbase + n) * ldk + kbase + 8 * lh;
  BF16Tile r;
  r.u4[0] = *(const uint4*)(p);
  r.u4[1] = *(const uint4*)(p + 16);
  return r.v;
}

// ---------------- conversion kernels ----------------

__global__ void cvt_x_bf16(const float* __restrict__ in, unsigned short* __restrict__ outp) {
  size_t i = (size_t)blockIdx.x * blockDim.x + threadIdx.x;
  if (i >= (size_t)T_TOK * H_DIM / 4) return;
  float4 v = ((const float4*)in)[i];
  unsigned int lo = (unsigned)f2bf(v.x) | ((unsigned)f2bf(v.y) << 16);
  unsigned int hi = (unsigned)f2bf(v.z) | ((unsigned)f2bf(v.w) << 16);
  ((uint2*)outp)[i] = make_uint2(lo, hi);
}

// [E][R][C] f32 -> [E][C][R] bf16, tiled LDS transpose (R,C multiples of 32)
__global__ void cvt_transpose_bf16(const float* __restrict__ in,
                                   unsigned short* __restrict__ outp,
                                   int R, int C) {
  __shared__ float tilebuf[32][33];
  const size_t eb = blockIdx.z;
  const float* ip = in + eb * (size_t)R * C;
  unsigned short* op = outp + eb * (size_t)R * C;
  const int cb = blockIdx.x * 32, rb = blockIdx.y * 32;
  const int tx = threadIdx.x, ty = threadIdx.y;   // blockDim (32,8)
#pragma unroll
  for (int j = 0; j < 32; j += 8)
    tilebuf[ty + j][tx] = ip[(size_t)(rb + ty + j) * C + cb + tx];
  __syncthreads();
#pragma unroll
  for (int j = 0; j < 32; j += 8)
    op[(size_t)(cb + ty + j) * R + rb + tx] = f2bf(tilebuf[tx][ty + j]);
}

// ---------------- router ----------------
__global__ void moe_router(const float* __restrict__ x, const float* __restrict__ Wg,
                           float* __restrict__ gate_w, unsigned int* __restrict__ list,
                           unsigned int* __restrict__ cnt, float* __restrict__ psum) {
  __shared__ float ps_l[E_EXP];
  const int tid = threadIdx.x;
  if (tid < E_EXP) ps_l[tid] = 0.f;
  __syncthreads();
  const int wave = tid >> 5, lane = tid & 31;
  const int t = blockIdx.x * 8 + wave;
  float acc[E_EXP];
#pragma unroll
  for (int e = 0; e < E_EXP; ++e) acc[e] = 0.f;
  const float* xr = x + (size_t)t * H_DIM;
  for (int h = lane; h < H_DIM; h += 32) {
    float xv = xr[h];
    const float* wr = Wg + (size_t)h * E_EXP;
#pragma unroll
    for (int e = 0; e < E_EXP; ++e) acc[e] = fmaf(xv, wr[e], acc[e]);
  }
#pragma unroll
  for (int e = 0; e < E_EXP; ++e) {
#pragma unroll
    for (int off = 16; off >= 1; off >>= 1)
      acc[e] += __shfl_xor(acc[e], off, 32);
  }
  if (lane == 0) {
    float mx = acc[0];
#pragma unroll
    for (int e = 1; e < E_EXP; ++e) mx = fmaxf(mx, acc[e]);
    float p[E_EXP], s = 0.f;
#pragma unroll
    for (int e = 0; e < E_EXP; ++e) { p[e] = __expf(acc[e] - mx); s += p[e]; }
    const float inv = 1.f / s;
#pragma unroll
    for (int e = 0; e < E_EXP; ++e) { p[e] *= inv; atomicAdd(&ps_l[e], p[e]); }
    int i0 = 0; float b0 = p[0];
#pragma unroll
    for (int e = 1; e < E_EXP; ++e) if (p[e] > b0) { b0 = p[e]; i0 = e; }
    int i1 = (i0 == 0) ? 1 : 0; float bsec = p[i1];
#pragma unroll
    for (int e = 0; e < E_EXP; ++e) if (e != i0 && p[e] > bsec) { bsec = p[e]; i1 = e; }
    const float wsum = b0 + bsec;
    gate_w[2 * t]     = b0   / wsum;
    gate_w[2 * t + 1] = bsec / wsum;
    unsigned int p0 = atomicAdd(&cnt[i0], 1u);
    list[(size_t)i0 * T_TOK + p0] = (unsigned)(2 * t);
    unsigned int p1 = atomicAdd(&cnt[i1], 1u);
    list[(size_t)i1 * T_TOK + p1] = (unsigned)(2 * t + 1);
  }
  __syncthreads();
  if (tid < E_EXP) atomicAdd(&psum[tid], ps_l[tid]);
}

// ---------------- fused expert FFN ----------------
// Block = 32-token tile of one expert's list; 8 waves (wave32).
// x tile (32 x 1024 bf16 = 64 KB) staged in LDS via async global->LDS DMA
// when available. Per 128-wide F chunk:
//   GEMM1: wave w computes h(32x16) for f-cols [fc+16w): one B load feeds
//          two WMMAs (both M-halves), K=1024 loop; silu -> hs[32][128].
//   GEMM2: wave w accumulates y(32x128) slice; each B load feeds 2 WMMAs.
// B-operand reuse across M-halves halves L2 weight traffic vs M=16.
__global__ __launch_bounds__(256) void moe_expert_ffn(
    const unsigned short* __restrict__ xb,
    const unsigned short* __restrict__ W1t,   // [E][F][H] bf16
    const unsigned short* __restrict__ W2t,   // [E][H][F] bf16
    const float* __restrict__ b1g,            // [E][F]
    const float* __restrict__ b2g,            // [E][H]
    const float* __restrict__ gate_w,
    const unsigned int* __restrict__ list,
    const unsigned int* __restrict__ cnt,
    float* __restrict__ out) {
  const int e = blockIdx.x & (E_EXP - 1);
  const int tile = blockIdx.x >> 4;
  const unsigned int ne = cnt[e];
  if ((unsigned)(tile * MTILE) >= ne) return;

  __shared__ __align__(16) unsigned short xs[MTILE * H_DIM];  // 64 KB
  __shared__ __align__(16) unsigned short hs[MTILE * 128];    // 8 KB
  __shared__ float wrow[MTILE];
  __shared__ unsigned int trow[MTILE];

  const int tid = threadIdx.x;
  const int lane = tid & 31;
  const int wv = tid >> 5;
  const int nlo = lane & 15;
  const int lh = lane >> 4;

  if (tid < MTILE) {
    unsigned int r = tile * MTILE + tid;
    if (r < ne) {
      unsigned int ent = list[(size_t)e * T_TOK + r];
      trow[tid] = ent >> 1;
      wrow[tid] = gate_w[ent];
    } else {  // pad: replicate first row's token, zero weight (skipped at store)
      unsigned int ent0 = list[(size_t)e * T_TOK + tile * MTILE];
      trow[tid] = ent0 >> 1;
      wrow[tid] = 0.f;
    }
  }
  __syncthreads();

  {  // gather x tile: 8 threads per row, 256B contiguous per thread
    const int rr = tid >> 3;   // 0..31
    const int cc = tid & 7;    // 8 x 256B chunks per 2048B row
    const unsigned short* srow = xb + (size_t)trow[rr] * H_DIM + cc * 128;
    unsigned short* drow = xs + rr * H_DIM + cc * 128;
#if __has_builtin(__builtin_amdgcn_global_load_async_to_lds_b128) && \
    __has_builtin(__builtin_amdgcn_s_wait_asynccnt)
#pragma unroll
    for (int j = 0; j < 16; ++j) {  // 16 x b128 async DMA per thread
      __builtin_amdgcn_global_load_async_to_lds_b128(
          (as1_v4i_p)(srow + j * 8), (as3_v4i_p)(drow + j * 8), 0, 0);
    }
    __builtin_amdgcn_s_wait_asynccnt(0);
#else
    const uint4* s4 = (const uint4*)srow;
    uint4* d4 = (uint4*)drow;
#pragma unroll
    for (int j = 0; j < 16; ++j) d4[j] = s4[j];
#endif
  }
  __syncthreads();

  // y accumulators: 2 M-halves x 8 N-tiles, init with b2 (gate-scaled at end)
  v8f yacc[2][8];
#pragma unroll
  for (int r2 = 0; r2 < 8; ++r2) {
    float bb = b2g[(size_t)e * H_DIM + wv * 128 + r2 * 16 + nlo];
#pragma unroll
    for (int q = 0; q < 8; ++q) { yacc[0][r2][q] = bb; yacc[1][r2][q] = bb; }
  }

  const unsigned short* W1e = W1t + (size_t)e * F_DIM * H_DIM;
  const unsigned short* W2e = W2t + (size_t)e * H_DIM * F_DIM;

  for (int fc = 0; fc < F_DIM; fc += 128) {
    if (fc + 128 < F_DIM)  // warm GL2 with next chunk's W1 columns
      __builtin_prefetch(W1e + (size_t)(fc + 128 + wv * 16) * H_DIM, 0, 1);

    // GEMM1: hidden pre-activation for this wave's 16 f-columns, both M-halves
    const int fcol = fc + wv * 16;
    v8f hc0, hc1;
    {
      float bb = b1g[(size_t)e * F_DIM + fcol + nlo];
#pragma unroll
      for (int q = 0; q < 8; ++q) { hc0[q] = bb; hc1[q] = bb; }
    }
    for (int kb = 0; kb < H_DIM; kb += 32) {
      v16bf b  = glbB(W1e, H_DIM, fcol, kb);
      v16bf a0 = ldsA(xs, H_DIM, kb);
      v16bf a1 = ldsA(xs + 16 * H_DIM, H_DIM, kb);
      hc0 = __builtin_amdgcn_wmma_f32_16x16x32_bf16(false, a0, false, b,
                                                    (short)0, hc0, false, false);
      hc1 = __builtin_amdgcn_wmma_f32_16x16x32_bf16(false, a1, false, b,
                                                    (short)0, hc1, false, false);
    }
    __syncthreads();  // prior chunk's GEMM2 reads of hs complete
#pragma unroll
    for (int q = 0; q < 8; ++q) {  // silu, pack to shared h tile (C map m=q+8lh)
      float v0 = hc0[q], v1 = hc1[q];
      hs[(q + 8 * lh) * 128 + wv * 16 + nlo]        = f2bf(v0 / (1.f + __expf(-v0)));
      hs[(q + 8 * lh + 16) * 128 + wv * 16 + nlo]   = f2bf(v1 / (1.f + __expf(-v1)));
    }
    __syncthreads();  // hs ready
    // GEMM2: y += h(32x128) @ W2[fc:fc+128, n-slice]; B reused across M-halves
#pragma unroll
    for (int kb2 = 0; kb2 < 128; kb2 += 32) {
      v16bf a0 = ldsA(hs, 128, kb2);
      v16bf a1 = ldsA(hs + 16 * 128, 128, kb2);
#pragma unroll
      for (int r2 = 0; r2 < 8; ++r2) {
        v16bf bm = glbB(W2e, F_DIM, wv * 128 + r2 * 16, fc + kb2);
        yacc[0][r2] = __builtin_amdgcn_wmma_f32_16x16x32_bf16(false, a0, false, bm,
                                                              (short)0, yacc[0][r2],
                                                              false, false);
        yacc[1][r2] = __builtin_amdgcn_wmma_f32_16x16x32_bf16(false, a1, false, bm,
                                                              (short)0, yacc[1][r2],
                                                              false, false);
      }
    }
  }

  // scatter: out[t, n] += gate_w * y  (global_atomic_add_f32)
#pragma unroll
  for (int mh = 0; mh < 2; ++mh) {
#pragma unroll
    for (int r2 = 0; r2 < 8; ++r2) {
#pragma unroll
      for (int q = 0; q < 8; ++q) {
        const int m = q + 8 * lh + 16 * mh;
        const float wgt = wrow[m];
        if (wgt != 0.f) {
          const int n = wv * 128 + r2 * 16 + nlo;
          atomicAdd(&out[(size_t)trow[m] * H_DIM + n], wgt * yacc[mh][r2][q]);
        }
      }
    }
  }
}

// aux = E * sum_e (cnt[e]/T) * (psum[e]/T) / K
__global__ void moe_aux(const unsigned int* __restrict__ cnt,
                        const float* __restrict__ psum,
                        float* __restrict__ aux_out) {
  if (threadIdx.x == 0 && blockIdx.x == 0) {
    float s = 0.f;
    for (int e = 0; e < E_EXP; ++e)
      s += ((float)cnt[e] / (float)T_TOK) * (psum[e] / (float)T_TOK);
    *aux_out = (float)E_EXP * s / (float)TOPK;
  }
}

extern "C" void kernel_launch(void* const* d_in, const int* in_sizes, int n_in,
                              void* d_out, int out_size, void* d_ws, size_t ws_size,
                              hipStream_t stream) {
  const float* x  = (const float*)d_in[0];
  const float* Wg = (const float*)d_in[1];
  const float* W1 = (const float*)d_in[2];
  const float* b1 = (const float*)d_in[3];
  const float* W2 = (const float*)d_in[4];
  const float* b2 = (const float*)d_in[5];
  float* out = (float*)d_out;

  // Workspace carve-out (~303 MB total), 256B aligned slices.
  char* ws = (char*)d_ws;
  size_t off = 0;
  auto carve = [&](size_t bytes) -> void* {
    void* p = ws + off;
    off += (bytes + 255) & ~(size_t)255;
    return p;
  };
  unsigned int* cnt  = (unsigned int*)carve(128);                 // cnt[16] + psum[16]
  float*        psum = (float*)(cnt + E_EXP);
  float*        gate = (float*)carve((size_t)T_TOK * 2 * 4);      // 128 KB
  unsigned int* list = (unsigned int*)carve((size_t)E_EXP * T_TOK * 4);      // 1 MB
  unsigned short* xb  = (unsigned short*)carve((size_t)T_TOK * H_DIM * 2);   // 32 MB
  unsigned short* W1t = (unsigned short*)carve((size_t)E_EXP * H_DIM * F_DIM * 2); // 128 MB
  unsigned short* W2t = (unsigned short*)carve((size_t)E_EXP * H_DIM * F_DIM * 2); // 128 MB
  (void)ws_size; (void)in_sizes; (void)n_in;

  // Zero routing counters and output accumulator (graph-capture safe).
  (void)hipMemsetAsync(cnt, 0, 128, stream);
  (void)hipMemsetAsync(out, 0, (size_t)out_size * sizeof(float), stream);

  // Precision conversion: x -> bf16; W1 [E][H][F] -> [E][F][H] bf16;
  // W2 [E][F][H] -> [E][H][F] bf16 (K-contiguous for WMMA B operand).
  cvt_x_bf16<<<(T_TOK * H_DIM / 4 + 255) / 256, 256, 0, stream>>>(x, xb);
  dim3 tb(32, 8);
  cvt_transpose_bf16<<<dim3(F_DIM / 32, H_DIM / 32, E_EXP), tb, 0, stream>>>(W1, W1t, H_DIM, F_DIM);
  cvt_transpose_bf16<<<dim3(H_DIM / 32, F_DIM / 32, E_EXP), tb, 0, stream>>>(W2, W2t, F_DIM, H_DIM);

  // Route, then sparse grouped FFN (one expert & 32-token tile per block),
  // then aux-loss finalize into out[T*H].
  moe_router<<<T_TOK / 8, 256, 0, stream>>>(x, Wg, gate, list, cnt, psum);
  moe_expert_ffn<<<E_EXP * (T_TOK / MTILE), 256, 0, stream>>>(xb, W1t, W2t, b1, b2,
                                                              gate, list, cnt, out);
  moe_aux<<<1, 32, 0, stream>>>(cnt, psum, out + (size_t)T_TOK * H_DIM);
}
